// ProbabilityFlow_61151744360563
// MI455X (gfx1250) — matrix operand order repaired and checked by reading, not compiled, not run on hardware
//
#include <hip/hip_runtime.h>

typedef __attribute__((ext_vector_type(16))) _Float16 v16h;
typedef __attribute__((ext_vector_type(8)))  _Float16 v8h;
typedef __attribute__((ext_vector_type(8)))  float    v8f;

#define B_N 8192
#define D_N 32
#define H_N 512
#define WAVES 4
#define NTHREADS (WAVES * 32)

// LDS: 2x W1^T (f16, [col][k] : 512x32) = 64KB, 2x W2^T (f16, [col][k] : 32x512) = 64KB,
// w1last/b1/cdiv (f32 x512 x2) = 12KB, b2, per-wave f16 staging tiles. ~144KB of 320KB/WGP.
struct __align__(16) Smem {
    _Float16 w1t[2][H_N][D_N];     // w1t[m][col][k]  = W1[k, col]   (k < 32)
    _Float16 w2t[2][D_N][H_N];     // w2t[m][col][k]  = W2[k, col]
    float    w1last[2][H_N];       // W1[32, col]  (t weights)
    float    b1[2][H_N];
    float    cdiv[2][H_N];         // c[k] = sum_i W1[i,k]*W2[k,i]
    float    b2[2][D_N];
    _Float16 hstage[WAVES][16][32];// per-wave gelu(z) staging tile for GEMM2 transpose
};

__device__ __forceinline__ float fast_tanh(float u) {
#if __has_builtin(__builtin_amdgcn_tanhf)
    return __builtin_amdgcn_tanhf(u);          // gfx1250 v_tanh_f32
#else
    // 1 - 2*rcp(exp(2u)+1): v_exp_f32 + v_rcp_f32, no IEEE divide sequence.
    // exp overflow -> +inf -> rcp -> 0 -> tanh -> 1; u << 0 -> rcp(1) -> -1. Robust.
    return 1.0f - 2.0f * __builtin_amdgcn_rcpf(__expf(2.0f * u) + 1.0f);
#endif
}

__device__ __forceinline__ v16h cat16(v8h lo, v8h hi) {
    return __builtin_shufflevector(lo, hi, 0,1,2,3,4,5,6,7,8,9,10,11,12,13,14,15);
}

__global__ __launch_bounds__(NTHREADS)
void ProbabilityFlow_kernel(const float* __restrict__ x,  const float* __restrict__ t,
                            const float* __restrict__ W1v, const float* __restrict__ b1v,
                            const float* __restrict__ W2v, const float* __restrict__ b2v,
                            const float* __restrict__ W1s, const float* __restrict__ b1s,
                            const float* __restrict__ W2s, const float* __restrict__ b2s,
                            float* __restrict__ out) {
    __shared__ Smem sm;
    const int tid = threadIdx.x;

    const float* W1[2] = {W1v, W1s};
    const float* W2[2] = {W2v, W2s};
    const float* B1[2] = {b1v, b1s};
    const float* B2[2] = {b2v, b2s};

    // ---------- stage weights into LDS (f16, transposed) ----------
    #pragma unroll
    for (int m = 0; m < 2; ++m) {
        for (int idx = tid; idx < D_N * H_N; idx += NTHREADS) {
            int i = idx >> 9, k = idx & (H_N - 1);            // W1 row-major [33,512]
            sm.w1t[m][k][i] = (_Float16)W1[m][idx];
        }
        for (int idx = tid; idx < H_N * D_N; idx += NTHREADS) {
            int k = idx >> 5, i = idx & (D_N - 1);            // W2 row-major [512,32]
            sm.w2t[m][i][k] = (_Float16)W2[m][idx];
        }
        for (int k = tid; k < H_N; k += NTHREADS) {
            sm.w1last[m][k] = W1[m][D_N * H_N + k];
            sm.b1[m][k]     = B1[m][k];
            float c = 0.0f;
            for (int i = 0; i < D_N; ++i)
                c += W1[m][i * H_N + k] * W2[m][k * D_N + i];
            sm.cdiv[m][k] = c;
        }
        if (tid < D_N) sm.b2[m][tid] = B2[m][tid];
    }
    __syncthreads();

    const int wave = tid >> 5;
    const int lane = tid & 31;
    const int hi   = lane >> 4;       // lane half: 0 -> rows 0..7 / K 0..7,16..23 etc.
    const int ln   = lane & 15;
    const int ksel = hi ? 8 : 0;
    const int rb   = (blockIdx.x * WAVES + wave) * 16;   // 16 batch rows per wave

    // ---------- A fragment: X tile [16 x 32], WMMA 16-bit A layout ----------
    const float* xr = x + (size_t)(rb + ln) * D_N;
    float4 xa = *(const float4*)(xr + ksel);
    float4 xb = *(const float4*)(xr + ksel + 4);
    float4 xc = *(const float4*)(xr + 16 + ksel);
    float4 xd = *(const float4*)(xr + 16 + ksel + 4);
    v16h a_x;
    a_x[0]=(_Float16)xa.x;  a_x[1]=(_Float16)xa.y;  a_x[2]=(_Float16)xa.z;  a_x[3]=(_Float16)xa.w;
    a_x[4]=(_Float16)xb.x;  a_x[5]=(_Float16)xb.y;  a_x[6]=(_Float16)xb.z;  a_x[7]=(_Float16)xb.w;
    a_x[8]=(_Float16)xc.x;  a_x[9]=(_Float16)xc.y;  a_x[10]=(_Float16)xc.z; a_x[11]=(_Float16)xc.w;
    a_x[12]=(_Float16)xd.x; a_x[13]=(_Float16)xd.y; a_x[14]=(_Float16)xd.z; a_x[15]=(_Float16)xd.w;

    // t for the 8 rows this lane owns in the C layout (M = 8*hi + j)
    float t_row[8];
    #pragma unroll
    for (int j = 0; j < 8; ++j) t_row[j] = t[rb + 8 * hi + j];

    v8f  outacc[2][2];
    float divacc[2][8];
    #pragma unroll
    for (int m = 0; m < 2; ++m) {
        #pragma unroll
        for (int o = 0; o < 2; ++o) outacc[m][o] = (v8f){};
        #pragma unroll
        for (int j = 0; j < 8; ++j) divacc[m][j] = 0.0f;
    }

    const float kA = 0.7978845608028654f;   // sqrt(2/pi)
    const float kB = 0.044715f;

    #pragma unroll
    for (int m = 0; m < 2; ++m) {
        for (int k0 = 0; k0 < H_N; k0 += 32) {          // hidden-dim chunks of 32
            // -------- GEMM1: z[:, k0+16n .. +15] = X @ W1 + (t*w1last + b1) --------
            #pragma unroll
            for (int n = 0; n < 2; ++n) {
                const int col = k0 + 16 * n + ln;
                v8h blo = *(const v8h*)&sm.w1t[m][col][hi * 8];
                v8h bhi = *(const v8h*)&sm.w1t[m][col][16 + hi * 8];
                v16h bfrag = cat16(blo, bhi);
                const float w1l = sm.w1last[m][col];
                const float bb  = sm.b1[m][col];
                const float cc  = sm.cdiv[m][col];
                v8f cin;
                #pragma unroll
                for (int j = 0; j < 8; ++j) cin[j] = t_row[j] * w1l + bb;

                v8f z = __builtin_amdgcn_wmma_f32_16x16x32_f16(
                            false, a_x, false, bfrag, (short)0, cin, false, false);

                // gelu + divergence contribution + staging for GEMM2 transpose
                #pragma unroll
                for (int j = 0; j < 8; ++j) {
                    float zz = z[j];
                    float u  = kA * (zz + kB * zz * zz * zz);
                    float th = fast_tanh(u);
                    float h  = 0.5f * zz * (1.0f + th);
                    float du = kA * (1.0f + 3.0f * kB * zz * zz);
                    float gp = 0.5f * (1.0f + th) + 0.5f * zz * (1.0f - th * th) * du;
                    divacc[m][j] += gp * cc;
                    sm.hstage[wave][8 * hi + j][16 * n + ln] = (_Float16)h;
                }
            }
            // wave-private LDS staging: order stores before transposed reads
            asm volatile("s_wait_dscnt 0" ::: "memory");

            // -------- GEMM2: out += gelu(z)[16 x 32chunk] @ W2[chunk x 32] --------
            v8h alo = *(const v8h*)&sm.hstage[wave][ln][ksel];
            v8h ahi = *(const v8h*)&sm.hstage[wave][ln][16 + ksel];
            v16h a2 = cat16(alo, ahi);
            #pragma unroll
            for (int o = 0; o < 2; ++o) {
                const int colo = 16 * o + ln;
                v8h wlo = *(const v8h*)&sm.w2t[m][colo][k0 + hi * 8];
                v8h whi = *(const v8h*)&sm.w2t[m][colo][k0 + 16 + hi * 8];
                v16h b2f = cat16(wlo, whi);
                outacc[m][o] = __builtin_amdgcn_wmma_f32_16x16x32_f16(
                                   false, a2, false, b2f, (short)0, outacc[m][o], false, false);
            }
        }
    }

    // ---------- reduce divergence across the 16 lanes of each half ----------
    #pragma unroll
    for (int off = 8; off >= 1; off >>= 1) {
        #pragma unroll
        for (int m = 0; m < 2; ++m)
            #pragma unroll
            for (int j = 0; j < 8; ++j)
                divacc[m][j] += __shfl_xor(divacc[m][j], off, 32);
    }

    // ---------- combine v - gg*s, store vf and -div ----------
    #pragma unroll
    for (int j = 0; j < 8; ++j) {
        const int   row = rb + 8 * hi + j;
        const float gg  = 0.5f * (1.0f - 2.0f * t_row[j]);
        #pragma unroll
        for (int o = 0; o < 2; ++o) {
            const int i = 16 * o + ln;
            float val = (outacc[0][o][j] + sm.b2[0][i]) - gg * (outacc[1][o][j] + sm.b2[1][i]);
            out[(size_t)row * D_N + i] = val;
        }
        if (ln == 0)
            out[(size_t)B_N * D_N + row] = -(divacc[0][j] - gg * divacc[1][j]);
    }
}

extern "C" void kernel_launch(void* const* d_in, const int* in_sizes, int n_in,
                              void* d_out, int out_size, void* d_ws, size_t ws_size,
                              hipStream_t stream) {
    (void)in_sizes; (void)n_in; (void)out_size; (void)d_ws; (void)ws_size;
    dim3 grid(B_N / (16 * WAVES));   // 128 workgroups x 4 waves x 16 rows = 8192
    ProbabilityFlow_kernel<<<grid, NTHREADS, 0, stream>>>(
        (const float*)d_in[0], (const float*)d_in[1],
        (const float*)d_in[2], (const float*)d_in[3],
        (const float*)d_in[4], (const float*)d_in[5],
        (const float*)d_in[6], (const float*)d_in[7],
        (const float*)d_in[8], (const float*)d_in[9],
        (float*)d_out);
}
